// MoEFFN_14594298871891
// MI455X (gfx1250) — matrix operand order repaired and checked by reading, not compiled, not run on hardware
//
#include <hip/hip_runtime.h>

// ---------------- constants ----------------
#define DM   1024      // d_model
#define DF   4096      // d_ff
#define NE   8         // experts
#define NT   8192      // tokens (B*S)
#define MT   32        // tokens per workgroup tile
#define FC   64        // f-chunk width
#define KP   128       // k-panel width (phase 1)
#define NP   256       // n-panel width (phase 2)

typedef __attribute__((ext_vector_type(16))) __bf16        v16bf;
typedef __attribute__((ext_vector_type(8)))  float         f32x8;
typedef __attribute__((ext_vector_type(4)))  unsigned int  u32x4;
typedef __attribute__((ext_vector_type(8)))  unsigned int  u32x8;

__device__ __forceinline__ unsigned short f2bf(float x) {
    unsigned int b = __float_as_uint(x);
    unsigned int r = b + 0x7FFFu + ((b >> 16) & 1u);   // RNE
    return (unsigned short)(r >> 16);
}

// A fragment (16x32 bf16, M x K): lane r=lane&15 holds row M=r.
// lanes 0-15: K = 0..7 and 16..23 ; lanes 16-31: K = 8..15 and 24..31.
template <int STRIDE>
__device__ __forceinline__ v16bf load_a_frag(const unsigned short (*arr)[STRIDE],
                                             int row0, int kbase, int lane) {
    int r = row0 + (lane & 15);
    int koff = (lane >> 4) * 8;
    union { v16bf v; u32x4 q[2]; } f;
    f.q[0] = *(const u32x4*)&arr[r][kbase + koff];
    f.q[1] = *(const u32x4*)&arr[r][kbase + koff + 16];
    return f.v;
}

// B fragment (32x16 bf16, K x N) from K-contiguous (transposed) LDS:
// lane n=lane&15 holds column N=n; lanes 0-15: K = 0..7 and 16..23 ; lanes 16-31: K = 8..15 and 24..31.
template <int STRIDE>
__device__ __forceinline__ v16bf load_b_frag(const unsigned short (*arr)[STRIDE],
                                             int n0, int kbase, int lane) {
    int n = n0 + (lane & 15);
    int koff = (lane >> 4) * 8;
    union { v16bf v; u32x4 q[2]; } f;
    f.q[0] = *(const u32x4*)&arr[n][kbase + koff];
    f.q[1] = *(const u32x4*)&arr[n][kbase + koff + 16];
    return f.v;
}

// ---------------- x -> bf16 ----------------
__global__ void moe_cvt_x(const float* __restrict__ x, unsigned short* __restrict__ xbf) {
    size_t i = ((size_t)blockIdx.x * blockDim.x + threadIdx.x) * 4;
    unsigned long long v =
        (unsigned long long)f2bf(x[i + 0])
      | ((unsigned long long)f2bf(x[i + 1]) << 16)
      | ((unsigned long long)f2bf(x[i + 2]) << 32)
      | ((unsigned long long)f2bf(x[i + 3]) << 48);
    *(unsigned long long*)(xbf + i) = v;
}

// ---------------- router: softmax + top-2 + gather lists ----------------
__global__ void moe_router(const float* __restrict__ x, const float* __restrict__ Wr,
                           int* __restrict__ cnt, float* __restrict__ psum,
                           int* __restrict__ idxA, float* __restrict__ gateA) {
    int wv   = (blockIdx.x * blockDim.x + threadIdx.x) >> 5;   // one wave per token
    int lane = threadIdx.x & 31;
    if (wv >= NT) return;
    const float* xr = x + (size_t)wv * DM;
    float acc[NE];
#pragma unroll
    for (int e = 0; e < NE; ++e) acc[e] = 0.f;
    for (int d = lane; d < DM; d += 32) {
        float xv = xr[d];
        const float* wrr = Wr + d * NE;
#pragma unroll
        for (int e = 0; e < NE; ++e) acc[e] += xv * wrr[e];
    }
#pragma unroll
    for (int off = 16; off; off >>= 1)
#pragma unroll
        for (int e = 0; e < NE; ++e) acc[e] += __shfl_xor(acc[e], off);
    float m = acc[0];
#pragma unroll
    for (int e = 1; e < NE; ++e) m = fmaxf(m, acc[e]);
    float pr[NE], s = 0.f;
#pragma unroll
    for (int e = 0; e < NE; ++e) { pr[e] = __expf(acc[e] - m); s += pr[e]; }
    float inv = 1.f / s;
#pragma unroll
    for (int e = 0; e < NE; ++e) pr[e] *= inv;
    int i1 = 0;
#pragma unroll
    for (int e = 1; e < NE; ++e) if (pr[e] > pr[i1]) i1 = e;
    int i2 = (i1 == 0) ? 1 : 0;
#pragma unroll
    for (int e = 0; e < NE; ++e) if (e != i1 && pr[e] > pr[i2]) i2 = e;
    float t = pr[i1] + pr[i2];
    if (lane == 0) {
        int p1 = atomicAdd(&cnt[i1], 1); idxA[i1 * NT + p1] = wv; gateA[i1 * NT + p1] = pr[i1] / t;
        int p2 = atomicAdd(&cnt[i2], 1); idxA[i2 * NT + p2] = wv; gateA[i2 * NT + p2] = pr[i2] / t;
    }
    if (lane < NE) atomicAdd(&psum[lane], pr[lane]);
}

// ---------------- aux loss ----------------
__global__ void moe_finalize(const int* __restrict__ cnt, const float* __restrict__ psum,
                             float* __restrict__ out_aux) {
    if (threadIdx.x == 0 && blockIdx.x == 0) {
        float a = 0.f;
        for (int e = 0; e < NE; ++e)
            a += ((float)cnt[e] / (float)NT) * (psum[e] / (float)NT);
        out_aux[0] = 0.01f * (float)NE * a;
    }
}

// ---------------- fused grouped SwiGLU FFN (WMMA bf16 + TDM gather) ----------------
__global__ __launch_bounds__(256)
void moe_ffn(const unsigned short* __restrict__ xbf,
             const float* __restrict__ Wgate, const float* __restrict__ Wup,
             const float* __restrict__ Wdown,
             const int* __restrict__ cnt, const int* __restrict__ idxA,
             const float* __restrict__ gateA, float* __restrict__ out) {
    __shared__ unsigned short Xs [MT][DM];        // 64 KB  x tile (bf16), TDM gather dest
    __shared__ unsigned short WgT[FC][KP + 8];    // 17 KB  Wgate panel, transposed [f][k]
    __shared__ unsigned short WuT[FC][KP + 8];    // 17 KB
    __shared__ unsigned short WdT[NP][FC + 8];    // 36 KB  Wdown panel, transposed [n][k]
    __shared__ unsigned short hS [MT][FC + 8];    // 4.5 KB h tile (bf16)
    __shared__ int   tokLds[MT];
    __shared__ float gateLds[MT];

    const int tid  = threadIdx.x;
    const int lane = tid & 31;
    const int wv   = tid >> 5;            // 8 waves
    const int msub = wv >> 2;             // 0..1  (M sub-tile of 16)
    const int grp  = wv & 3;              // 0..3  (f sub-tile / n group)

    const int e    = blockIdx.x >> 8;
    const int mt   = blockIdx.x & 255;
    const int cntE = cnt[e];
    const int rbase = mt * MT;
    if (rbase >= cntE) return;            // uniform for whole block

    if (tid < MT) {
        int r = rbase + tid;
        if (r < cntE) { tokLds[tid] = idxA[e * NT + r]; gateLds[tid] = gateA[e * NT + r]; }
        else          { tokLds[tid] = -1;               gateLds[tid] = 0.f; }
    }
    __syncthreads();

    // ---- stage x tile via Tensor Data Mover, gather mode (16-bit row indices) ----
    // Two descriptors: rows 0-15 and 16-31. Invalid rows use index 0; their output
    // rows are skipped (tok < 0) in the epilogue, so the data never escapes.
    if (wv == 0) {
        unsigned ldsb = (unsigned)(unsigned long long)&Xs[0][0];
        unsigned long long ga = (unsigned long long)xbf;
#pragma unroll
        for (int half = 0; half < 2; ++half) {
            u32x4 g0; u32x8 g1; u32x4 g2, g3;
            // group0: count=1 | gather_index_size=16b(bit30=0) | gather_mode(bit31)
            g0[0] = 0x80000001u;
            g0[1] = ldsb + (unsigned)(half * 16 * DM * 2);        // lds_addr
            g0[2] = (unsigned)ga;                                 // global_addr[31:0]
            g0[3] = ((unsigned)(ga >> 32) & 0x01FFFFFFu) | 0x80000000u; // addr[56:32] | type=2
            // group1: wg_mask=0, data_size=2B, no pad/iterate/atomic-barrier
            g1[0] = 1u << 16;                 // data_size = 1 (2 bytes)
            g1[1] = (unsigned)DM << 16;       // tensor_dim0[15:0]
            g1[2] = (unsigned)NT << 16;       // tensor_dim0[31:16]=0 | tensor_dim1[15:0]
            g1[3] = (unsigned)DM << 16;       // tensor_dim1[31:16]=0 | tile_dim0
            g1[4] = 16u;                      // tile_dim1 = #valid gather indices
            g1[5] = (unsigned)DM;             // tensor_dim0_stride[31:0]
            g1[6] = 0u;
            g1[7] = 0u;
            // groups 2+3: 16 packed 16-bit row indices (token ids)
            unsigned w[8];
#pragma unroll
            for (int j = 0; j < 8; ++j) {
                int ta = tokLds[half * 16 + 2 * j];     if (ta < 0) ta = 0;
                int tb = tokLds[half * 16 + 2 * j + 1]; if (tb < 0) tb = 0;
                w[j] = (unsigned)__builtin_amdgcn_readfirstlane(ta | (tb << 16));
            }
            g2[0] = w[0]; g2[1] = w[1]; g2[2] = w[2]; g2[3] = w[3];
            g3[0] = w[4]; g3[1] = w[5]; g3[2] = w[6]; g3[3] = w[7];
            asm volatile("tensor_load_to_lds %0, %1, %2, %3"
                         :: "s"(g0), "s"(g1), "s"(g2), "s"(g3)
                         : "memory");
        }
        __builtin_amdgcn_s_wait_tensorcnt(0);
    }

    const float* Wg = Wgate + (size_t)e * DM * DF;
    const float* Wu = Wup   + (size_t)e * DM * DF;
    const float* Wd = Wdown + (size_t)e * DF * DM;

    const f32x8 zero8 = {};
    f32x8 accY[16];                        // wave's 16 output tiles, persistent
#pragma unroll
    for (int i = 0; i < 16; ++i) accY[i] = zero8;

    for (int fc = 0; fc < DF / FC; ++fc) {
        const int f0 = fc * FC;
        f32x8 accG = zero8, accU = zero8;

        // ---- phase 1: g = x@Wg, u = x@Wu over K=1024 in panels of 128 ----
        for (int kp = 0; kp < DM / KP; ++kp) {
            __syncthreads();
            for (int i = tid; i < FC * KP; i += 256) {
                int n = i & (FC - 1), k = i >> 6;
                size_t gidx = (size_t)(kp * KP + k) * DF + f0 + n;
                WgT[n][k] = f2bf(Wg[gidx]);
                WuT[n][k] = f2bf(Wu[gidx]);
            }
            if (kp + 1 < DM / KP)
                __builtin_prefetch(Wg + (size_t)((kp + 1) * KP) * DF + f0 + (tid & 63), 0, 1);
            __syncthreads();
#pragma unroll
            for (int ks = 0; ks < KP / 32; ++ks) {
                v16bf a  = load_a_frag(Xs,  msub * 16, kp * KP + ks * 32, lane);
                v16bf bg = load_b_frag(WgT, grp * 16,  ks * 32, lane);
                v16bf bu = load_b_frag(WuT, grp * 16,  ks * 32, lane);
                accG = __builtin_amdgcn_wmma_f32_16x16x32_bf16(false, a, false, bg,
                                                               (short)0, accG, false, false);
                accU = __builtin_amdgcn_wmma_f32_16x16x32_bf16(false, a, false, bu,
                                                               (short)0, accU, false, false);
            }
        }

        // ---- SiLU(g) * u -> hS (C/D layout: VGPR v => M=v+8*(lane>=16), N=lane&15) ----
        __syncthreads();
        {
            int nn = grp * 16 + (lane & 15);
            int mb = msub * 16 + ((lane >> 4) << 3);
#pragma unroll
            for (int v = 0; v < 8; ++v) {
                float g = accG[v];
                float h = (g / (1.f + __expf(-g))) * accU[v];
                hS[mb + v][nn] = f2bf(h);
            }
        }
        __syncthreads();

        // ---- phase 2: y += h @ Wd, N=1024 in panels of 256 ----
#pragma unroll
        for (int p = 0; p < DM / NP; ++p) {
            __syncthreads();
            for (int i = tid; i < NP * FC; i += 256) {
                int n = i & (NP - 1), k = i >> 8;
                WdT[n][k] = f2bf(Wd[(size_t)(f0 + k) * DM + p * NP + n]);
            }
            __syncthreads();
#pragma unroll
            for (int ks = 0; ks < FC / 32; ++ks) {
                v16bf a = load_a_frag(hS, msub * 16, ks * 32, lane);
#pragma unroll
                for (int j = 0; j < 4; ++j) {
                    v16bf b = load_b_frag(WdT, (grp * 4 + j) * 16, ks * 32, lane);
                    accY[p * 4 + j] = __builtin_amdgcn_wmma_f32_16x16x32_bf16(
                        false, a, false, b, (short)0, accY[p * 4 + j], false, false);
                }
            }
        }
    }

    // ---- epilogue: gated scatter-add (each token receives exactly 2 expert adds) ----
#pragma unroll
    for (int p = 0; p < 4; ++p)
#pragma unroll
        for (int j = 0; j < 4; ++j) {
            int ncol = p * NP + (grp * 4 + j) * 16 + (lane & 15);
#pragma unroll
            for (int v = 0; v < 8; ++v) {
                int m = msub * 16 + v + ((lane >> 4) << 3);
                int tok = tokLds[m];
                if (tok >= 0)
                    atomicAdd(out + (size_t)tok * DM + ncol, accY[p * 4 + j][v] * gateLds[m]);
            }
        }
}

// ---------------- host ----------------
extern "C" void kernel_launch(void* const* d_in, const int* in_sizes, int n_in,
                              void* d_out, int out_size, void* d_ws, size_t ws_size,
                              hipStream_t stream) {
    const float* x  = (const float*)d_in[0];
    const float* Wr = (const float*)d_in[1];
    const float* Wg = (const float*)d_in[2];
    const float* Wu = (const float*)d_in[3];
    const float* Wd = (const float*)d_in[4];
    float* out = (float*)d_out;

    // workspace layout
    unsigned short* xbf = (unsigned short*)d_ws;                     // NT*DM bf16 = 16 MB
    char* hdr    = (char*)d_ws + (size_t)NT * DM * 2;
    int*   cnt   = (int*)hdr;                                        // 8 ints
    float* psum  = (float*)(hdr + 32);                               // 8 floats
    int*   idxA  = (int*)(hdr + 64);                                 // NE*NT ints
    float* gateA = (float*)(hdr + 64 + (size_t)NE * NT * 4);         // NE*NT floats

    hipMemsetAsync(d_out, 0, (size_t)(NT * DM + 1) * sizeof(float), stream);
    hipMemsetAsync(hdr, 0, 64, stream);

    moe_cvt_x <<<(NT * DM / 4) / 256, 256, 0, stream>>>(x, xbf);
    moe_router<<<NT / 8, 256, 0, stream>>>(x, Wr, cnt, psum, idxA, gateA);
    moe_ffn   <<<NE * (NT / MT), 256, 0, stream>>>(xbf, Wg, Wu, Wd, cnt, idxA, gateA, out);
    moe_finalize<<<1, 32, 0, stream>>>(cnt, psum, out + (size_t)NT * DM);
}
